// ExpertHead_61838939128386
// MI455X (gfx1250) — matrix operand order repaired and checked by reading, not compiled, not run on hardware
//
#include <hip/hip_runtime.h>
#include <hip/hip_bf16.h>
#include <math.h>

#define D     512
#define SEQ   4096
#define BATCH 4
#define KT    64   // key block for flash attention

typedef __attribute__((ext_vector_type(16))) __bf16 bf16x16;
typedef __attribute__((ext_vector_type(8)))  __bf16 bf16x8;
typedef __attribute__((ext_vector_type(8)))  float  f32x8;
typedef __attribute__((ext_vector_type(8)))  unsigned short us8;

union Frag { bf16x16 v; bf16x8 h[2]; };

__device__ __forceinline__ unsigned short f2bf(float f) {
  union { float f; unsigned int u; } x; x.f = f;
  unsigned int u = x.u;
  u += 0x7FFFu + ((u >> 16) & 1u);   // round-to-nearest-even
  return (unsigned short)(u >> 16);
}

// gfx1250 async global->LDS copy (ASYNCcnt-tracked), ISA §15.18.3 op 98.
__device__ __forceinline__ void async_g2lds_b128(const void* gaddr, unsigned lds_addr) {
  asm volatile("global_load_async_to_lds_b128 %0, %1, off"
               :: "v"(lds_addr), "v"(gaddr) : "memory");
}
__device__ __forceinline__ void wait_async0() {
  asm volatile("s_wait_asynccnt 0x0" ::: "memory");
}

// ---------------------------------------------------------------- convert
__global__ void cvt_f32_bf16(const float* __restrict__ src,
                             unsigned short* __restrict__ dst, int n) {
  int i = blockIdx.x * blockDim.x + threadIdx.x;
  if (i < n) dst[i] = f2bf(src[i]);
}

// ---------------------------------------------------------------- QKV GEMM
// C[16384 x 512] = xb @ W^T + b ; blockIdx.z selects Q/K/V; Q pre-scaled.
// Software-pipelined: next k-step's fragments load while current WMMAs run.
__global__ __launch_bounds__(128)
void qkv_gemm(const unsigned short* __restrict__ xb,
              const unsigned short* __restrict__ Wq,
              const unsigned short* __restrict__ Wk,
              const unsigned short* __restrict__ Wv,
              const float* __restrict__ bq, const float* __restrict__ bk,
              const float* __restrict__ bv,
              unsigned short* __restrict__ Qs, unsigned short* __restrict__ Kb,
              unsigned short* __restrict__ Vb) {
  const int wave = threadIdx.x >> 5, lane = threadIdx.x & 31;
  const int g = lane >> 4, ln = lane & 15;
  const int row0 = blockIdx.x * 64 + wave * 16;
  const int n0   = blockIdx.y * 64;
  const int which = blockIdx.z;
  const unsigned short* W   = which == 0 ? Wq : (which == 1 ? Wk : Wv);
  const float*          bia = which == 0 ? bq : (which == 1 ? bk : bv);
  unsigned short*       out = which == 0 ? Qs : (which == 1 ? Kb : Vb);
  const float scale = which == 0 ? 0.044194173824159216f : 1.0f; // 1/sqrt(512)

  f32x8 acc[4];
  for (int t = 0; t < 4; ++t)
    for (int i = 0; i < 8; ++i) acc[t][i] = 0.0f;

  const unsigned short* arow = xb + (size_t)(row0 + ln) * D;
  const unsigned short* brow[4];
#pragma unroll
  for (int t = 0; t < 4; ++t)
    brow[t] = W + (size_t)(n0 + t * 16 + ln) * D;

  Frag a[2], b[2][4];
  auto loadK = [&](int p, int kb) {
    a[p].h[0] = *(const bf16x8*)(arow + kb + 16 * g);
    a[p].h[1] = *(const bf16x8*)(arow + kb + 16 * g + 8);
#pragma unroll
    for (int t = 0; t < 4; ++t) {
      b[p][t].h[0] = *(const bf16x8*)(brow[t] + kb + 16 * g);
      b[p][t].h[1] = *(const bf16x8*)(brow[t] + kb + 16 * g + 8);
    }
  };

  loadK(0, 0);
  int p = 0;
  for (int kb = 0; kb < D; kb += 32, p ^= 1) {
    if (kb + 32 < D) loadK(p ^ 1, kb + 32);
#pragma unroll
    for (int t = 0; t < 4; ++t)
      acc[t] = __builtin_amdgcn_wmma_f32_16x16x32_bf16(
          false, a[p].v, false, b[p][t].v, (short)0, acc[t], false, false);
  }
#pragma unroll
  for (int t = 0; t < 4; ++t) {
    int col = n0 + t * 16 + ln;
    float bb = bia[col];
#pragma unroll
    for (int i = 0; i < 8; ++i) {
      int r = row0 + i + 8 * g;
      out[(size_t)r * D + col] = f2bf((acc[t][i] + bb) * scale);
    }
  }
}

// ---------------------------------------------------------------- attention
// One block = 16 query rows of one batch. 4 waves: wave w scores keys
// [w*16,w*16+16) (full-D reduction), then owns D-slice [w*128,(w+1)*128)
// of the flash-attention O accumulator. LN + exact GELU fused at the end.
// Q/K tiles staged with GLOBAL_LOAD_ASYNC_TO_LDS_B128; V transposed manually.
__global__ __launch_bounds__(128)
void attn_kernel(const unsigned short* __restrict__ Q,
                 const unsigned short* __restrict__ K,
                 const unsigned short* __restrict__ V,
                 const float* __restrict__ ln_g, const float* __restrict__ ln_b,
                 unsigned short* __restrict__ Y) {
  __shared__ __align__(16) unsigned short Ktile[KT * D];   // 64KB (aliased as Obuf later)
  __shared__ __align__(16) unsigned short Vt[D * KT];      // 64KB transposed V
  __shared__ __align__(16) unsigned short Qt[16 * D];      // 16KB
  __shared__ __align__(16) float scoreBuf[16 * KT];        // 4KB
  __shared__ __align__(16) unsigned short Pb[16 * KT];     // 2KB
  __shared__ float mrow[16], lrow[16], arow[16], muL[16], rvL[16];

  const int tid = threadIdx.x;
  const int wave = tid >> 5, lane = tid & 31;
  const int g = lane >> 4, ln = lane & 15;
  const int q0 = blockIdx.x * 16;
  const size_t base = (size_t)blockIdx.y * SEQ * D;
  const unsigned short* Qp = Q + base + (size_t)q0 * D;
  const unsigned short* Kp = K + base;
  const unsigned short* Vp = V + base;

  const unsigned QtA = (unsigned)(size_t)(void*)Qt;     // LDS byte addresses
  const unsigned KtA = (unsigned)(size_t)(void*)Ktile;

  // Q tile: async global->LDS (rows contiguous in global)
  for (int idx = tid; idx < 16 * D / 8; idx += 128)
    async_g2lds_b128(Qp + idx * 8, QtA + idx * 16);
  if (tid < 16) { mrow[tid] = -1e30f; lrow[tid] = 0.0f; }

  f32x8 o[8];
  for (int t = 0; t < 8; ++t)
    for (int i = 0; i < 8; ++i) o[t][i] = 0.0f;

  for (int kb = 0; kb < SEQ; kb += KT) {
    __syncthreads();                       // tiles free (and mrow ready)
    // K tile: async DMA straight to LDS (row-major, 64 rows contiguous)
    for (int idx = tid; idx < KT * D / 8; idx += 128)
      async_g2lds_b128(Kp + (size_t)kb * D + idx * 8, KtA + idx * 16);
    // V tile: load + transpose into Vt
    for (int idx = tid; idx < KT * D / 8; idx += 128) {
      int r = idx >> 6;                    // D/8 == 64 chunks per row
      int c = (idx & 63) << 3;
      us8 v = *(const us8*)(Vp + (size_t)(kb + r) * D + c);
#pragma unroll
      for (int j = 0; j < 8; ++j) Vt[(c + j) * KT + r] = v[j];
    }
    wait_async0();                         // Q (first iter) + K asyncs landed
    __syncthreads();

    // ---- scores: S_w[16q x 16k] = Q(16xD) @ K_w^T
    f32x8 s;
    for (int i = 0; i < 8; ++i) s[i] = 0.0f;
    const unsigned short* krow = Ktile + (size_t)(wave * 16 + ln) * D;
    const unsigned short* qrow = Qt + (size_t)ln * D;
    for (int kk = 0; kk < D; kk += 32) {
      Frag a, bfr;
      a.h[0]   = *(const bf16x8*)(qrow + kk + 16 * g);
      a.h[1]   = *(const bf16x8*)(qrow + kk + 16 * g + 8);
      bfr.h[0] = *(const bf16x8*)(krow + kk + 16 * g);
      bfr.h[1] = *(const bf16x8*)(krow + kk + 16 * g + 8);
      s = __builtin_amdgcn_wmma_f32_16x16x32_bf16(
          false, a.v, false, bfr.v, (short)0, s, false, false);
    }
#pragma unroll
    for (int i = 0; i < 8; ++i)
      scoreBuf[(i + 8 * g) * KT + wave * 16 + ln] = s[i];
    __syncthreads();

    // ---- online softmax stats (one thread per query row)
    if (tid < 16) {
      int q = tid;
      float mb = -1e30f;
      for (int k = 0; k < KT; ++k) mb = fmaxf(mb, scoreBuf[q * KT + k]);
      float m_old = mrow[q];
      float m_new = fmaxf(m_old, mb);
      float alpha = expf(m_old - m_new);
      float sum = 0.0f;
      for (int k = 0; k < KT; ++k) {
        float p = expf(scoreBuf[q * KT + k] - m_new);
        Pb[q * KT + k] = f2bf(p);
        sum += p;
      }
      lrow[q] = lrow[q] * alpha + sum;
      mrow[q] = m_new;
      arow[q] = alpha;
    }
    __syncthreads();

    // ---- O_w = O_w*alpha + P(16x64) @ V[:, wave-slice]
    float al[8];
#pragma unroll
    for (int i = 0; i < 8; ++i) al[i] = arow[i + 8 * g];
    Frag pa[2];
    const unsigned short* prow = Pb + ln * KT;
#pragma unroll
    for (int ss = 0; ss < 2; ++ss) {
      pa[ss].h[0] = *(const bf16x8*)(prow + 32 * ss + 16 * g);
      pa[ss].h[1] = *(const bf16x8*)(prow + 32 * ss + 16 * g + 8);
    }
#pragma unroll
    for (int t = 0; t < 8; ++t) {
#pragma unroll
      for (int i = 0; i < 8; ++i) o[t][i] *= al[i];
      const unsigned short* vcol = Vt + (wave * 128 + t * 16 + ln) * KT;
      Frag b0, b1;
      b0.h[0] = *(const bf16x8*)(vcol + 16 * g);
      b0.h[1] = *(const bf16x8*)(vcol + 16 * g + 8);
      b1.h[0] = *(const bf16x8*)(vcol + 32 + 16 * g);
      b1.h[1] = *(const bf16x8*)(vcol + 32 + 16 * g + 8);
      o[t] = __builtin_amdgcn_wmma_f32_16x16x32_bf16(
          false, pa[0].v, false, b0.v, (short)0, o[t], false, false);
      o[t] = __builtin_amdgcn_wmma_f32_16x16x32_bf16(
          false, pa[1].v, false, b1.v, (short)0, o[t], false, false);
    }
  }
  __syncthreads();

  // ---- 1/l, stage O in LDS (alias Ktile), LayerNorm + exact GELU
  float* Obuf = (float*)Ktile;
  float li[8];
#pragma unroll
  for (int i = 0; i < 8; ++i) li[i] = 1.0f / lrow[i + 8 * g];
#pragma unroll
  for (int t = 0; t < 8; ++t) {
    int dcol = wave * 128 + t * 16 + ln;
#pragma unroll
    for (int i = 0; i < 8; ++i) Obuf[(i + 8 * g) * D + dcol] = o[t][i] * li[i];
  }
  __syncthreads();
  if (tid < 16) {
    int q = tid;
    float sum = 0.0f, sq = 0.0f;
    for (int d = 0; d < D; ++d) { float v = Obuf[q * D + d]; sum += v; sq += v * v; }
    float mu = sum * (1.0f / D);
    float var = sq * (1.0f / D) - mu * mu;
    muL[q] = mu;
    rvL[q] = rsqrtf(var + 1e-5f);
  }
  __syncthreads();
  for (int idx = tid; idx < 16 * D; idx += 128) {
    int q = idx >> 9, d = idx & (D - 1);
    float v = (Obuf[idx] - muL[q]) * rvL[q] * ln_g[d] + ln_b[d];
    float ge = 0.5f * v * (1.0f + erff(v * 0.70710678118654752f));
    Y[base + (size_t)(q0 + q) * D + d] = f2bf(ge);
  }
}

// ---------------------------------------------------------------- out proj
__global__ __launch_bounds__(128)
void out_gemm(const unsigned short* __restrict__ Yb,
              const unsigned short* __restrict__ Wo,
              const float* __restrict__ bo, float* __restrict__ out) {
  const int wave = threadIdx.x >> 5, lane = threadIdx.x & 31;
  const int g = lane >> 4, ln = lane & 15;
  const int row0 = blockIdx.x * 64 + wave * 16;
  const int n0   = blockIdx.y * 64;

  f32x8 acc[4];
  for (int t = 0; t < 4; ++t)
    for (int i = 0; i < 8; ++i) acc[t][i] = 0.0f;

  const unsigned short* arow = Yb + (size_t)(row0 + ln) * D;
  const unsigned short* brow[4];
#pragma unroll
  for (int t = 0; t < 4; ++t)
    brow[t] = Wo + (size_t)(n0 + t * 16 + ln) * D;

  Frag a[2], b[2][4];
  auto loadK = [&](int p, int kb) {
    a[p].h[0] = *(const bf16x8*)(arow + kb + 16 * g);
    a[p].h[1] = *(const bf16x8*)(arow + kb + 16 * g + 8);
#pragma unroll
    for (int t = 0; t < 4; ++t) {
      b[p][t].h[0] = *(const bf16x8*)(brow[t] + kb + 16 * g);
      b[p][t].h[1] = *(const bf16x8*)(brow[t] + kb + 16 * g + 8);
    }
  };

  loadK(0, 0);
  int p = 0;
  for (int kb = 0; kb < D; kb += 32, p ^= 1) {
    if (kb + 32 < D) loadK(p ^ 1, kb + 32);
#pragma unroll
    for (int t = 0; t < 4; ++t)
      acc[t] = __builtin_amdgcn_wmma_f32_16x16x32_bf16(
          false, a[p].v, false, b[p][t].v, (short)0, acc[t], false, false);
  }
#pragma unroll
  for (int t = 0; t < 4; ++t) {
    int col = n0 + t * 16 + ln;
    float bb = bo[col];
#pragma unroll
    for (int i = 0; i < 8; ++i) {
      int r = row0 + i + 8 * g;
      out[(size_t)r * D + col] = acc[t][i] + bb;
    }
  }
}

// ---------------------------------------------------------------- launch
extern "C" void kernel_launch(void* const* d_in, const int* in_sizes, int n_in,
                              void* d_out, int out_size, void* d_ws, size_t ws_size,
                              hipStream_t stream) {
  (void)in_sizes; (void)n_in; (void)out_size; (void)ws_size;
  const float* x    = (const float*)d_in[0];
  const float* Wq   = (const float*)d_in[1];
  const float* bq   = (const float*)d_in[2];
  const float* Wk   = (const float*)d_in[3];
  const float* bk   = (const float*)d_in[4];
  const float* Wv   = (const float*)d_in[5];
  const float* bv   = (const float*)d_in[6];
  const float* ln_g = (const float*)d_in[7];
  const float* ln_b = (const float*)d_in[8];
  const float* Wo   = (const float*)d_in[9];
  const float* bo   = (const float*)d_in[10];
  float* out = (float*)d_out;

  const size_t NACT = (size_t)BATCH * SEQ * D;   // 8,388,608
  const size_t NW   = (size_t)D * D;             // 262,144
  unsigned short* xb  = (unsigned short*)d_ws;
  unsigned short* Qs  = xb + NACT;
  unsigned short* Kb  = Qs + NACT;
  unsigned short* Vb  = Kb + NACT;
  unsigned short* Yb  = Vb + NACT;
  unsigned short* Wqb = Yb + NACT;
  unsigned short* Wkb = Wqb + NW;
  unsigned short* Wvb = Wkb + NW;
  unsigned short* Wob = Wvb + NW;

  hipLaunchKernelGGL(cvt_f32_bf16, dim3((NACT + 255) / 256), dim3(256), 0, stream, x,  xb,  (int)NACT);
  hipLaunchKernelGGL(cvt_f32_bf16, dim3((NW + 255) / 256),   dim3(256), 0, stream, Wq, Wqb, (int)NW);
  hipLaunchKernelGGL(cvt_f32_bf16, dim3((NW + 255) / 256),   dim3(256), 0, stream, Wk, Wkb, (int)NW);
  hipLaunchKernelGGL(cvt_f32_bf16, dim3((NW + 255) / 256),   dim3(256), 0, stream, Wv, Wvb, (int)NW);
  hipLaunchKernelGGL(cvt_f32_bf16, dim3((NW + 255) / 256),   dim3(256), 0, stream, Wo, Wob, (int)NW);

  hipLaunchKernelGGL(qkv_gemm, dim3(BATCH * SEQ / 64, D / 64, 3), dim3(128), 0, stream,
                     xb, Wqb, Wkb, Wvb, bq, bk, bv, Qs, Kb, Vb);

  hipLaunchKernelGGL(attn_kernel, dim3(SEQ / 16, BATCH), dim3(128), 0, stream,
                     Qs, Kb, Vb, ln_g, ln_b, Yb);

  hipLaunchKernelGGL(out_gemm, dim3(BATCH * SEQ / 64, D / 64), dim3(128), 0, stream,
                     Yb, Wob, bo, out);
}
